// TopKGate_41592463294490
// MI455X (gfx1250) — 8-bit matrix operand order repaired and checked by reading, compiled, not run on hardware
//
#include <hip/hip_runtime.h>
#include <hip/hip_bf16.h>
#include <math.h>

typedef __attribute__((ext_vector_type(8))) int v8i;

#define DCONST 4096
#define ECONST 64
#define KTOP   8
#define EPSQ   1e-5f

__global__ __launch_bounds__(1024)
void wquant_kernel(const float* __restrict__ W, int* __restrict__ wq32,
                   float* __restrict__ wdq, int N  ) {
    __shared__ float red[1024];
    __shared__ float s_sw;
    const int tid = threadIdx.x;
    float s = 0.f;
    for (int i = tid; i < N; i += 1024) s += fabsf(W[i]);
    red[tid] = s;
    __syncthreads();
    for (int off = 512; off > 0; off >>= 1) {
        if (tid < off) red[tid] += red[tid + off];
        __syncthreads();
    }
    if (tid == 0) {
        float mean = red[0] / (float)N;
        float dq = fmaxf(mean, EPSQ);
        *wdq = dq;
        s_sw = 1.0f / dq;
    }
    __syncthreads();
    const float sw = s_sw;

    const int nDwords = N >> 2;
    for (int o = tid; o < nDwords; o += 1024) {
        const int d    = o & 7;
        const int lane = (o >> 3) & 31;
        const int kc   = (o >> 8) & 63;
        const int nt   = o >> 14;
        const int e    = nt * 16 + (lane & 15);
        const int k    = (kc << 6) + ((d >> 2) << 5) + ((d & 3) << 2)
                       + ((lane >> 4) << 4);
        const float* wp = W + (size_t)e * DCONST + k;
        int pack = 0;
        #pragma unroll
        for (int j = 0; j < 4; ++j) {
            float q = rintf(wp[j] * sw);
            q = fminf(fmaxf(q, -1.0f), 1.0f);
            pack |= ((int)q & 255) << (8 * j);
        }
        wq32[o] = pack;
    }
}

__global__ __launch_bounds__(256)
void actquant_kernel(const float* __restrict__ x, int* __restrict__ xq32,
                     float* __restrict__ sx_arr, int T, int D) {
    __shared__ float lds_sx[16];

    const int wave   = threadIdx.x >> 5;
    const int lane   = threadIdx.x & 31;
    const int lane15 = lane & 15;
    const int hi     = lane >> 4;
    const int tile   = blockIdx.x;

    {
        const int lrow = wave * 2 + hi;
        const int row  = tile * 16 + lrow;
        const float4* xr = (const float4*)(x + (size_t)row * D);
        const int n4 = D >> 2;
        float m = 0.f;
        for (int i = lane15; i < n4; i += 16) {
            float4 v = xr[i];
            m = fmaxf(m, fmaxf(fmaxf(fabsf(v.x), fabsf(v.y)),
                               fmaxf(fabsf(v.z), fabsf(v.w))));
        }
        #pragma unroll
        for (int off = 8; off > 0; off >>= 1)
            m = fmaxf(m, __shfl_xor(m, off, 32));
        const float sx = 127.0f / fmaxf(m, EPSQ);
        if (lane15 == 0) {
            lds_sx[lrow] = sx;
            sx_arr[row]  = sx;
        }
    }
    __syncthreads();

    const float sxa = lds_sx[lane15];
    const float* xrow = x + (size_t)(tile * 16 + lane15) * D;
    #pragma unroll
    for (int i = 0; i < 8; ++i) {
        const int kc    = wave * 8 + i;
        const int kbase = kc << 6;
        v8i a;
        #pragma unroll
        for (int d = 0; d < 8; ++d) {
            const int koff = kbase + ((d >> 2) << 5) + ((d & 3) << 2) + (hi << 4);
            float4 v = *(const float4*)(xrow + koff);
            int q0 = (int)fminf(fmaxf(rintf(v.x * sxa), -128.f), 127.f);
            int q1 = (int)fminf(fmaxf(rintf(v.y * sxa), -128.f), 127.f);
            int q2 = (int)fminf(fmaxf(rintf(v.z * sxa), -128.f), 127.f);
            int q3 = (int)fminf(fmaxf(rintf(v.w * sxa), -128.f), 127.f);
            a[d] = (q0 & 255) | ((q1 & 255) << 8) | ((q2 & 255) << 16)
                 | ((q3 & 255) << 24);
        }
        v8i* dst = (v8i*)(xq32 + ((size_t)((size_t)tile * 64 + kc) * 32 + lane) * 8);
        *dst = a;
    }
}

__global__ __launch_bounds__(256)
void moe_gate_kernel(const int* __restrict__ xq32,
                     const int* __restrict__ wq32,
                     const float* __restrict__ sx_arr,
                     const float* __restrict__ wdq,
                     float* __restrict__ w_out, int* __restrict__ i_out,
                     int T, int D) {
    __shared__ float lds_logits[8][16][ECONST + 1];

    const int wave   = threadIdx.x >> 5;
    const int lane   = threadIdx.x & 31;
    const int lane15 = lane & 15;
    const int hi     = lane >> 4;
    const int tile   = blockIdx.x * 8 + wave;
    const int rowBase = tile * 16;

    const float wdqs = *wdq;

    const v8i* aptr = (const v8i*)xq32 + ((size_t)tile * 64) * 32 + lane;
    const v8i* bptr = (const v8i*)wq32 + lane;

    v8i acc[4];
    #pragma unroll
    for (int nt = 0; nt < 4; ++nt) acc[nt] = (v8i){0, 0, 0, 0, 0, 0, 0, 0};

    const int nK = D >> 6;
    for (int kc = 0; kc < nK; ++kc) {
        if (kc + 1 < nK)
            __builtin_prefetch((const void*)(aptr + (size_t)(kc + 1) * 32), 0, 1);

        const v8i a = aptr[(size_t)kc * 32];

        #pragma unroll
        for (int nt = 0; nt < 4; ++nt) {
            const v8i b = bptr[(size_t)(nt * 64 + kc) * 32];
            acc[nt] = __builtin_amdgcn_wmma_i32_16x16x64_iu8(
                 true, a,  true, b, acc[nt],
                 false,  false);
        }
    }

    #pragma unroll
    for (int nt = 0; nt < 4; ++nt) {
        #pragma unroll
        for (int r = 0; r < 8; ++r) {
            const int row_l = r + (hi << 3);
            lds_logits[wave][row_l][nt * 16 + lane15] = (float)acc[nt][r] * wdqs;
        }
    }
    __syncthreads();

    if (lane < 16) {
        const int row = rowBase + lane;
        const float f = 1.0f / sx_arr[row];
        float* lg = lds_logits[wave][lane];
        for (int c = 0; c < ECONST; ++c) lg[c] *= f;

        float topv[KTOP];
        int   topi[KTOP];
        #pragma unroll
        for (int j = 0; j < KTOP; ++j) {
            float best = -INFINITY; int bi = 0;
            for (int c = 0; c < ECONST; ++c) {
                float v = lg[c];
                if (v > best) { best = v; bi = c; }
            }
            topv[j] = best; topi[j] = bi;
            lg[bi] = -INFINITY;
        }

        float ew[KTOP], sum = 0.f;
        #pragma unroll
        for (int j = 0; j < KTOP; ++j) { ew[j] = __expf(topv[j] - topv[0]); sum += ew[j]; }
        const float inv = 1.0f / sum;

        float4  z = make_float4(0.f, 0.f, 0.f, 0.f);
        float4* orow = (float4*)(w_out + (size_t)row * ECONST);
        #pragma unroll
        for (int q = 0; q < ECONST / 4; ++q) orow[q] = z;
        #pragma unroll
        for (int j = 0; j < KTOP; ++j)
            w_out[(size_t)row * ECONST + topi[j]] = ew[j] * inv;
        #pragma unroll
        for (int j = 0; j < KTOP; ++j)
            i_out[(size_t)row * KTOP + j] = topi[j];
    }
}

extern "C" void kernel_launch(void* const* d_in, const int* in_sizes, int n_in,
                              void* d_out, int out_size, void* d_ws, size_t ws_size,
                              hipStream_t stream) {
    (void)n_in; (void)out_size; (void)ws_size;
    const float* x = (const float*)d_in[0];
    const float* W = (const float*)d_in[1];
    const int D = DCONST;
    const int E = ECONST;
    const int T = in_sizes[0] / D;

    char* ws = (char*)d_ws;
    int*   wq32 = (int*)ws;
    float* wdq  = (float*)(ws + (size_t)E * D);
    float* sxA  = (float*)(ws + (size_t)E * D + 256);
    size_t off_xq = ((size_t)E * D + 256 + (size_t)T * 4 + 1023) & ~(size_t)1023;
    int*   xq32 = (int*)(ws + off_xq);

    float* w_out = (float*)d_out;
    int*   i_out = (int*)d_out + (size_t)T * E;

    wquant_kernel<<<1, 1024, 0, stream>>>(W, wq32, wdq, E * D);
    actquant_kernel<<<T / 16, 256, 0, stream>>>(x, xq32, sxA, T, D);
    moe_gate_kernel<<<T / 128, 256, 0, stream>>>(xq32, wq32, sxA, wdq,
                                                 w_out, i_out, T, D);
}
